// Head_77309411328085
// MI455X (gfx1250) — matrix operand order repaired and checked by reading, not compiled, not run on hardware
//
#include <hip/hip_runtime.h>
#include <hip/hip_bf16.h>

// ---------------------------------------------------------------------------
// Attention head: out = softmax(Q Kᵀ / sqrt(C)) V, with Q/K/V = x @ W{q,k,v}ᵀ
// B=32, S=1024, C=D=512. All matmuls via v_wmma_f32_16x16x32_f16 (wave32).
// Block tile 128x256, 8 waves, 64x64 per wave -> 16 WMMA per K-step per wave.
// ---------------------------------------------------------------------------

typedef __attribute__((ext_vector_type(16))) _Float16 v16h;
typedef __attribute__((ext_vector_type(8)))  _Float16 v8h;
typedef __attribute__((ext_vector_type(8)))  float    v8f;
typedef __attribute__((ext_vector_type(4)))  float    v4f;

#define S_LEN   1024
#define DMODEL  512
#define NBATCH  32
#define BM      128
#define BN      256
#define BK      32
#define PITCH   48          // halves; 96 bytes/row, keeps every b128 16B aligned
#define NTHR    256

// workspace layout (bytes)
#define WH_BYTES   (3u * DMODEL * DMODEL * 2u)                 // 1.5 MB f16 weights
#define QKV_OFF    (WH_BYTES)
#define QKV_ONE    ((size_t)NBATCH * S_LEN * DMODEL * 2u)      // 32 MB per Q/K/V
#define S_OFF      (QKV_OFF + 3u * QKV_ONE)                    // fp32 scores, 128 MB
#define WSEL_H     ((size_t)NBATCH * S_LEN * DMODEL)           // half-stride per Q/K/V

// ---- LDS fragment loaders (ISA §7.12.2 layouts) ---------------------------
__device__ __forceinline__ v16h ld_frag_a(const _Float16* lds, int rowBase, int lane) {
    const int rr = rowBase + (lane & 15);
    const int k0 = (lane & 16) ? 8 : 0;
    const _Float16* p = lds + rr * PITCH + k0;
    v8h lo = *(const v8h*)(p);        // K = k0 .. k0+7
    v8h hi = *(const v8h*)(p + 16);   // K = k0+16 .. k0+23
    return __builtin_shufflevector(lo, hi, 0,1,2,3,4,5,6,7,8,9,10,11,12,13,14,15);
}
__device__ __forceinline__ v16h ld_frag_b(const _Float16* lds, int colBase, int lane) {
    const int rr = colBase + (lane & 15);
    const int k0 = (lane & 16) ? 8 : 0;
    const _Float16* p = lds + rr * PITCH + k0;
    v8h lo = *(const v8h*)(p);
    v8h hi = *(const v8h*)(p + 16);
    return __builtin_shufflevector(lo, hi, 0,1,2,3,4,5,6,7,8,9,10,11,12,13,14,15);
}

// ---- kernel 1: fp32 weights -> f16 ----------------------------------------
__global__ __launch_bounds__(NTHR) void convert_w(const float* __restrict__ Wq,
                                                  const float* __restrict__ Wk,
                                                  const float* __restrict__ Wv,
                                                  _Float16* __restrict__ Wh) {
    const float* src = (blockIdx.y == 0) ? Wq : (blockIdx.y == 1) ? Wk : Wv;
    const int i = blockIdx.x * NTHR + threadIdx.x;                 // 0..262143
    Wh[(size_t)blockIdx.y * (DMODEL * DMODEL) + i] = (_Float16)src[i];
}

// ---- kernel 2: QKV projection  (NT: out[s,d] = sum_c x[s,c] W[d,c]) -------
__global__ __launch_bounds__(NTHR) void qkv_gemm(const float* __restrict__ x,
                                                 const _Float16* __restrict__ Wh,
                                                 _Float16* __restrict__ qkv) {
    __shared__ _Float16 ldsA[BM * PITCH];   // 12 KB
    __shared__ _Float16 ldsB[BN * PITCH];   // 24 KB
    const int b    = blockIdx.z;
    const int wsel = blockIdx.y >> 1;               // 0=Q 1=K 2=V
    const int n0   = (blockIdx.y & 1) * BN;
    const int m0   = blockIdx.x * BM;
    const float*    A   = x  + (size_t)b * S_LEN * DMODEL;
    const _Float16* Bm  = Wh + (size_t)wsel * DMODEL * DMODEL;
    _Float16*       Out = qkv + (size_t)wsel * WSEL_H + (size_t)b * S_LEN * DMODEL;

    const int tid = threadIdx.x, lane = tid & 31, wave = tid >> 5;
    const int wm = (wave & 1) * 64, wn = (wave >> 1) * 64;
    const int r = tid >> 1, h = (tid & 1) * 16;     // A staging map

    v8f acc[4][4];
#pragma unroll
    for (int mi = 0; mi < 4; ++mi)
#pragma unroll
        for (int ni = 0; ni < 4; ++ni) { v8f z = {0,0,0,0,0,0,0,0}; acc[mi][ni] = z; }

    for (int kk = 0; kk < DMODEL; kk += BK) {
        { // stage A (fp32 -> f16), 2 threads per row
            const float* src = A + (size_t)(m0 + r) * DMODEL + kk + h;
            v4f f0 = *(const v4f*)(src);
            v4f f1 = *(const v4f*)(src + 4);
            v4f f2 = *(const v4f*)(src + 8);
            v4f f3 = *(const v4f*)(src + 12);
            v8h p0, p1;
#pragma unroll
            for (int i = 0; i < 4; ++i) { p0[i] = (_Float16)f0[i]; p0[i+4] = (_Float16)f1[i];
                                          p1[i] = (_Float16)f2[i]; p1[i+4] = (_Float16)f3[i]; }
            _Float16* dst = &ldsA[r * PITCH + h];
            *(v8h*)dst = p0; *(v8h*)(dst + 8) = p1;
            if (kk + BK < DMODEL) __builtin_prefetch(src + BK, 0, 3);
        }
        { // stage B (f16 copy), 1 thread per row of 32 halves
            const _Float16* src = Bm + (size_t)(n0 + tid) * DMODEL + kk;
            v8h q0 = *(const v8h*)(src);
            v8h q1 = *(const v8h*)(src + 8);
            v8h q2 = *(const v8h*)(src + 16);
            v8h q3 = *(const v8h*)(src + 24);
            _Float16* dst = &ldsB[tid * PITCH];
            *(v8h*)dst = q0; *(v8h*)(dst + 8) = q1;
            *(v8h*)(dst + 16) = q2; *(v8h*)(dst + 24) = q3;
            if (kk + BK < DMODEL) __builtin_prefetch(src + BK, 0, 3);
        }
        __syncthreads();
        v16h af[4], bf[4];
#pragma unroll
        for (int mi = 0; mi < 4; ++mi) af[mi] = ld_frag_a(ldsA, wm + mi * 16, lane);
#pragma unroll
        for (int ni = 0; ni < 4; ++ni) bf[ni] = ld_frag_b(ldsB, wn + ni * 16, lane);
#pragma unroll
        for (int mi = 0; mi < 4; ++mi)
#pragma unroll
            for (int ni = 0; ni < 4; ++ni)
                acc[mi][ni] = __builtin_amdgcn_wmma_f32_16x16x32_f16(
                    false, af[mi], false, bf[ni], (short)0, acc[mi][ni], false, false);
        __syncthreads();
    }
    const int nCol = lane & 15, mRow = (lane & 16) ? 8 : 0;
#pragma unroll
    for (int mi = 0; mi < 4; ++mi)
#pragma unroll
        for (int ni = 0; ni < 4; ++ni)
#pragma unroll
            for (int i = 0; i < 8; ++i)
                Out[(size_t)(m0 + wm + mi * 16 + mRow + i) * DMODEL
                    + n0 + wn + ni * 16 + nCol] = (_Float16)acc[mi][ni][i];
}

// ---- kernel 3: scores S[s,t] = (1/sqrt(C)) * sum_d Q[s,d] K[t,d] ----------
__global__ __launch_bounds__(NTHR) void scores_gemm(const _Float16* __restrict__ Qh,
                                                    const _Float16* __restrict__ Kh,
                                                    float* __restrict__ Sf) {
    __shared__ _Float16 ldsA[BM * PITCH];
    __shared__ _Float16 ldsB[BN * PITCH];
    const int b  = blockIdx.z;
    const int n0 = blockIdx.y * BN;                 // t-block (grid.y = 4)
    const int m0 = blockIdx.x * BM;
    const _Float16* A   = Qh + (size_t)b * S_LEN * DMODEL;
    const _Float16* Bm  = Kh + (size_t)b * S_LEN * DMODEL;
    float*          Out = Sf + (size_t)b * S_LEN * S_LEN;

    const int tid = threadIdx.x, lane = tid & 31, wave = tid >> 5;
    const int wm = (wave & 1) * 64, wn = (wave >> 1) * 64;
    const int r = tid >> 1, h = (tid & 1) * 16;

    v8f acc[4][4];
#pragma unroll
    for (int mi = 0; mi < 4; ++mi)
#pragma unroll
        for (int ni = 0; ni < 4; ++ni) { v8f z = {0,0,0,0,0,0,0,0}; acc[mi][ni] = z; }

    for (int kk = 0; kk < DMODEL; kk += BK) {
        { // stage Q rows
            const _Float16* src = A + (size_t)(m0 + r) * DMODEL + kk + h;
            v8h p0 = *(const v8h*)src, p1 = *(const v8h*)(src + 8);
            _Float16* dst = &ldsA[r * PITCH + h];
            *(v8h*)dst = p0; *(v8h*)(dst + 8) = p1;
            if (kk + BK < DMODEL) __builtin_prefetch(src + BK, 0, 3);
        }
        { // stage K rows
            const _Float16* src = Bm + (size_t)(n0 + tid) * DMODEL + kk;
            v8h q0 = *(const v8h*)(src);
            v8h q1 = *(const v8h*)(src + 8);
            v8h q2 = *(const v8h*)(src + 16);
            v8h q3 = *(const v8h*)(src + 24);
            _Float16* dst = &ldsB[tid * PITCH];
            *(v8h*)dst = q0; *(v8h*)(dst + 8) = q1;
            *(v8h*)(dst + 16) = q2; *(v8h*)(dst + 24) = q3;
            if (kk + BK < DMODEL) __builtin_prefetch(src + BK, 0, 3);
        }
        __syncthreads();
        v16h af[4], bf[4];
#pragma unroll
        for (int mi = 0; mi < 4; ++mi) af[mi] = ld_frag_a(ldsA, wm + mi * 16, lane);
#pragma unroll
        for (int ni = 0; ni < 4; ++ni) bf[ni] = ld_frag_b(ldsB, wn + ni * 16, lane);
#pragma unroll
        for (int mi = 0; mi < 4; ++mi)
#pragma unroll
            for (int ni = 0; ni < 4; ++ni)
                acc[mi][ni] = __builtin_amdgcn_wmma_f32_16x16x32_f16(
                    false, af[mi], false, bf[ni], (short)0, acc[mi][ni], false, false);
        __syncthreads();
    }
    const float scale = 0.04419417382415922f;  // 1/sqrt(512)
    const int nCol = lane & 15, mRow = (lane & 16) ? 8 : 0;
#pragma unroll
    for (int mi = 0; mi < 4; ++mi)
#pragma unroll
        for (int ni = 0; ni < 4; ++ni)
#pragma unroll
            for (int i = 0; i < 8; ++i)
                Out[(size_t)(m0 + wm + mi * 16 + mRow + i) * S_LEN
                    + n0 + wn + ni * 16 + nCol] = acc[mi][ni][i] * scale;
}

// ---- kernel 4: row softmax, fp32 in, f16 probs written in place -----------
// P row r lives in the first 2048 bytes of the row's 4096-byte fp32 slot
// (half pitch = 2048). All loads complete before the first barrier, all
// stores happen after the last one, so the in-place overwrite is safe.
__global__ __launch_bounds__(NTHR) void softmax_rows(float* __restrict__ Sf) {
    const size_t row = blockIdx.x;                      // 0..32767
    const float* p  = Sf + row * S_LEN;
    _Float16*   ph  = (_Float16*)Sf + row * 2048;
    const int tid = threadIdx.x, lane = tid & 31, wave = tid >> 5;
    __shared__ float red_m[8], red_s[8];

    float v[4];
#pragma unroll
    for (int j = 0; j < 4; ++j) v[j] = p[tid + j * NTHR];

    float m = fmaxf(fmaxf(v[0], v[1]), fmaxf(v[2], v[3]));
#pragma unroll
    for (int off = 16; off >= 1; off >>= 1) m = fmaxf(m, __shfl_xor(m, off, 32));
    if (lane == 0) red_m[wave] = m;
    __syncthreads();
    m = red_m[0];
#pragma unroll
    for (int w = 1; w < 8; ++w) m = fmaxf(m, red_m[w]);

    float e[4], s = 0.f;
#pragma unroll
    for (int j = 0; j < 4; ++j) { e[j] = __expf(v[j] - m); s += e[j]; }
#pragma unroll
    for (int off = 16; off >= 1; off >>= 1) s += __shfl_xor(s, off, 32);
    if (lane == 0) red_s[wave] = s;
    __syncthreads();
    s = red_s[0];
#pragma unroll
    for (int w = 1; w < 8; ++w) s += red_s[w];
    const float inv = 1.0f / s;
#pragma unroll
    for (int j = 0; j < 4; ++j) ph[tid + j * NTHR] = (_Float16)(e[j] * inv);
}

// ---- kernel 5: out[s,d] = sum_t P[s,t] V[t,d]  (NN; V staged transposed) --
__global__ __launch_bounds__(NTHR) void pv_gemm(const _Float16* __restrict__ Ph,
                                                const _Float16* __restrict__ Vh,
                                                float* __restrict__ Out0) {
    __shared__ _Float16 ldsA[BM * PITCH];
    __shared__ _Float16 ldsB[BN * PITCH];
    const int b  = blockIdx.z;
    const int n0 = blockIdx.y * BN;                 // d-block (grid.y = 2)
    const int m0 = blockIdx.x * BM;                 // s-block
    const _Float16* A   = Ph + (size_t)b * S_LEN * 2048;   // pitch 2048 halves
    const _Float16* Vm  = Vh + (size_t)b * S_LEN * DMODEL;
    float*          Out = Out0 + (size_t)b * S_LEN * DMODEL;

    const int tid = threadIdx.x, lane = tid & 31, wave = tid >> 5;
    const int wm = (wave & 1) * 64, wn = (wave >> 1) * 64;
    const int r = tid >> 1, h = (tid & 1) * 16;     // A staging map
    const int tr = tid >> 3, dOff = (tid & 7) * 32; // V staging map (transpose)

    v8f acc[4][4];
#pragma unroll
    for (int mi = 0; mi < 4; ++mi)
#pragma unroll
        for (int ni = 0; ni < 4; ++ni) { v8f z = {0,0,0,0,0,0,0,0}; acc[mi][ni] = z; }

    for (int kk = 0; kk < S_LEN; kk += BK) {
        { // stage P rows (K = t contiguous)
            const _Float16* src = A + (size_t)(m0 + r) * 2048 + kk + h;
            v8h p0 = *(const v8h*)src, p1 = *(const v8h*)(src + 8);
            _Float16* dst = &ldsA[r * PITCH + h];
            *(v8h*)dst = p0; *(v8h*)(dst + 8) = p1;
            if (kk + BK < S_LEN) __builtin_prefetch(src + BK, 0, 3);
        }
        { // stage V transposed: ldsB[d][t] = V[t][d]
            const _Float16* src = Vm + (size_t)(kk + tr) * DMODEL + n0 + dOff;
            v8h q0 = *(const v8h*)(src);
            v8h q1 = *(const v8h*)(src + 8);
            v8h q2 = *(const v8h*)(src + 16);
            v8h q3 = *(const v8h*)(src + 24);
#pragma unroll
            for (int i = 0; i < 8; ++i) {
                ldsB[(dOff + i)      * PITCH + tr] = q0[i];
                ldsB[(dOff + 8 + i)  * PITCH + tr] = q1[i];
                ldsB[(dOff + 16 + i) * PITCH + tr] = q2[i];
                ldsB[(dOff + 24 + i) * PITCH + tr] = q3[i];
            }
            if (kk + BK < S_LEN) __builtin_prefetch(src + (size_t)BK * DMODEL, 0, 3);
        }
        __syncthreads();
        v16h af[4], bf[4];
#pragma unroll
        for (int mi = 0; mi < 4; ++mi) af[mi] = ld_frag_a(ldsA, wm + mi * 16, lane);
#pragma unroll
        for (int ni = 0; ni < 4; ++ni) bf[ni] = ld_frag_b(ldsB, wn + ni * 16, lane);
#pragma unroll
        for (int mi = 0; mi < 4; ++mi)
#pragma unroll
            for (int ni = 0; ni < 4; ++ni)
                acc[mi][ni] = __builtin_amdgcn_wmma_f32_16x16x32_f16(
                    false, af[mi], false, bf[ni], (short)0, acc[mi][ni], false, false);
        __syncthreads();
    }
    const int nCol = lane & 15, mRow = (lane & 16) ? 8 : 0;
#pragma unroll
    for (int mi = 0; mi < 4; ++mi)
#pragma unroll
        for (int ni = 0; ni < 4; ++ni)
#pragma unroll
            for (int i = 0; i < 8; ++i)
                Out[(size_t)(m0 + wm + mi * 16 + mRow + i) * DMODEL
                    + n0 + wn + ni * 16 + nCol] = acc[mi][ni][i];
}

// ---------------------------------------------------------------------------
extern "C" void kernel_launch(void* const* d_in, const int* in_sizes, int n_in,
                              void* d_out, int out_size, void* d_ws, size_t ws_size,
                              hipStream_t stream) {
    const float* x  = (const float*)d_in[0];
    const float* Wq = (const float*)d_in[1];
    const float* Wk = (const float*)d_in[2];
    const float* Wv = (const float*)d_in[3];
    float* out = (float*)d_out;

    char* ws = (char*)d_ws;
    _Float16* Wh  = (_Float16*)ws;               // 3 x 512 x 512 f16
    _Float16* QKV = (_Float16*)(ws + QKV_OFF);   // Q | K | V, 32x1024x512 f16 each
    float*    Sf  = (float*)(ws + S_OFF);        // 32x1024x1024 fp32 (P f16 in place)

    const _Float16* Qh = QKV;
    const _Float16* Kh = QKV + WSEL_H;
    const _Float16* Vh = QKV + 2 * WSEL_H;

    convert_w   <<<dim3(1024, 3, 1),  NTHR, 0, stream>>>(Wq, Wk, Wv, Wh);
    qkv_gemm    <<<dim3(8, 6, 32),    NTHR, 0, stream>>>(x, Wh, QKV);
    scores_gemm <<<dim3(8, 4, 32),    NTHR, 0, stream>>>(Qh, Kh, Sf);
    softmax_rows<<<dim3(32768, 1, 1), NTHR, 0, stream>>>(Sf);
    pv_gemm     <<<dim3(8, 2, 32),    NTHR, 0, stream>>>((const _Float16*)Sf, Vh, out);
}
